// SwapCountLoss_816043786446
// MI455X (gfx1250) — compile-verified, matrix operand order checked by reading, not run on hardware
//
#include <hip/hip_runtime.h>

typedef _Float16 v16h __attribute__((ext_vector_type(16)));
typedef _Float16 v8h  __attribute__((ext_vector_type(8)));
typedef float    v8f  __attribute__((ext_vector_type(8)));
typedef float    v4f  __attribute__((ext_vector_type(4)));

#define B_  32
#define N_  512
#define Q_  512
#define E_  4096

// ---------------------------------------------------------------------------
// Kernel 1: build d_swap = 3*max(d_hw-1,0) as f16, zero the per-batch accums.
// ---------------------------------------------------------------------------
__global__ void __launch_bounds__(256)
prep_kernel(const float* __restrict__ d_hw, _Float16* __restrict__ Dswap,
            float* __restrict__ accum /* 2*B floats */) {
    int idx = blockIdx.x * blockDim.x + threadIdx.x;
    if (idx < Q_ * Q_) {
        float v = d_hw[idx];
        Dswap[idx] = (_Float16)(3.0f * fmaxf(v - 1.0f, 0.0f));
    }
    if (idx < 2 * B_) accum[idx] = 0.0f;
}

// ---------------------------------------------------------------------------
// Kernel 2: T[b] = P[b] @ Dswap   (512x512 @ 512x512, f16 in, f32 accum, f16 out)
// Block = 128 threads (4 waves). Tile 64(M) x 64(N), K-step 32.
// Dswap is symmetric, so B-tile column loads are row loads.
// ---------------------------------------------------------------------------
__global__ void __launch_bounds__(128)
gemm_kernel(const float* __restrict__ P, const _Float16* __restrict__ Dswap,
            _Float16* __restrict__ T) {
    __shared__ __align__(16) _Float16 As[64][32];  // P tile [m][k]
    __shared__ __align__(16) _Float16 Bs[64][32];  // D tile [n][k] (via symmetry)

    const int b   = blockIdx.z;
    const int m0  = blockIdx.y * 64;
    const int c0  = blockIdx.x * 64;

    const int tid   = threadIdx.x;       // 0..127
    const int wave  = tid >> 5;          // 0..3
    const int lane  = tid & 31;
    const int mrow  = lane & 15;         // fragment row / col within 16
    const int khalf = lane >> 4;         // which 8-wide K half-group

    const float* Pb = P + (size_t)b * N_ * Q_;

    v8f acc[4] = {};

    for (int k0 = 0; k0 < Q_; k0 += 32) {
        __syncthreads();  // protect previous iteration's fragment reads
        {
            // stage tiles: 64x32 elems each, 128 threads -> one half-row each
            int row  = tid >> 1;
            int colh = (tid & 1) * 16;
            const float* src = Pb + (size_t)(m0 + row) * Q_ + k0 + colh;
#pragma unroll
            for (int i = 0; i < 4; i++) {
                v4f v = *(const v4f*)(src + 4 * i);
                As[row][colh + 4 * i + 0] = (_Float16)v[0];
                As[row][colh + 4 * i + 1] = (_Float16)v[1];
                As[row][colh + 4 * i + 2] = (_Float16)v[2];
                As[row][colh + 4 * i + 3] = (_Float16)v[3];
            }
            // Bs[n][k] = Dswap[c0+n][k0+k]  (== Dswap[k0+k][c0+n] by symmetry)
            const _Float16* dsrc = Dswap + (size_t)(c0 + row) * Q_ + k0 + colh;
            *(v8h*)&Bs[row][colh + 0] = *(const v8h*)(dsrc + 0);
            *(v8h*)&Bs[row][colh + 8] = *(const v8h*)(dsrc + 8);
        }
        __syncthreads();

        // A fragment: lane holds row (wave*16+mrow), K in {8*khalf..}+{16+8*khalf..}
        v8h alo = *(const v8h*)&As[wave * 16 + mrow][khalf * 8];
        v8h ahi = *(const v8h*)&As[wave * 16 + mrow][16 + khalf * 8];
        v16h a;
#pragma unroll
        for (int i = 0; i < 8; i++) { a[i] = alo[i]; a[8 + i] = ahi[i]; }

#pragma unroll
        for (int ns = 0; ns < 4; ns++) {
            v8h blo = *(const v8h*)&Bs[ns * 16 + mrow][khalf * 8];
            v8h bhi = *(const v8h*)&Bs[ns * 16 + mrow][16 + khalf * 8];
            v16h bb;
#pragma unroll
            for (int i = 0; i < 8; i++) { bb[i] = blo[i]; bb[8 + i] = bhi[i]; }
            acc[ns] = __builtin_amdgcn_wmma_f32_16x16x32_f16(
                false, a, false, bb, (short)0, acc[ns], false, false);
        }
    }

    // C/D layout: VGPR r, lane L -> M = r + 8*(L>>4), N = L&15
    _Float16* Tb = T + (size_t)b * N_ * Q_;
    const int rowb = m0 + wave * 16 + 8 * khalf;
    const int col  = lane & 15;
#pragma unroll
    for (int ns = 0; ns < 4; ns++) {
#pragma unroll
        for (int r = 0; r < 8; r++) {
            Tb[(size_t)(rowb + r) * Q_ + c0 + ns * 16 + col] = (_Float16)acc[ns][r];
        }
    }
}

// ---------------------------------------------------------------------------
// Kernel 3: per-edge cost = T[b][src] . P[b][dst]; accumulate w*cost and w
// per batch. One wave per edge, 8 edges (one batch) per 256-thread block.
// ---------------------------------------------------------------------------
__global__ void __launch_bounds__(256)
edge_kernel(const _Float16* __restrict__ T, const float* __restrict__ P,
            const int* __restrict__ esrc, const int* __restrict__ edst,
            const float* __restrict__ ew, float* __restrict__ accum) {
    const int blocksPerBatch = E_ / 8;                 // 512
    const int b     = blockIdx.x / blocksPerBatch;
    const int chunk = blockIdx.x % blocksPerBatch;
    const int wave  = threadIdx.x >> 5;
    const int lane  = threadIdx.x & 31;
    const int e     = chunk * 8 + wave;

    const int   s = esrc[b * E_ + e];
    const int   d = edst[b * E_ + e];
    const float w = ew[b * E_ + e];

    const _Float16* trow = T + ((size_t)b * N_ + s) * Q_;
    const float*    prow = P + ((size_t)b * N_ + d) * Q_;

    float sum = 0.0f;
    const int q0 = lane * 16;  // 32 lanes x 16 elems = 512
#pragma unroll
    for (int i = 0; i < 2; i++) {
        v8h tv = *(const v8h*)(trow + q0 + 8 * i);
        v4f p0 = *(const v4f*)(prow + q0 + 8 * i);
        v4f p1 = *(const v4f*)(prow + q0 + 8 * i + 4);
#pragma unroll
        for (int j = 0; j < 4; j++) {
            sum += (float)tv[j]     * p0[j];
            sum += (float)tv[4 + j] * p1[j];
        }
    }
    // wave32 reduction
#pragma unroll
    for (int off = 16; off > 0; off >>= 1)
        sum += __shfl_xor(sum, off, 32);

    __shared__ float snum[8];
    __shared__ float sden[8];
    if (lane == 0) { snum[wave] = w * sum; sden[wave] = w; }
    __syncthreads();
    if (threadIdx.x == 0) {
        float nsum = 0.0f, dsum = 0.0f;
#pragma unroll
        for (int i = 0; i < 8; i++) { nsum += snum[i]; dsum += sden[i]; }
        atomicAdd(&accum[b], nsum);
        atomicAdd(&accum[B_ + b], dsum);
    }
}

// ---------------------------------------------------------------------------
// Kernel 4: loss = mean_b( num[b] / max(den[b],1e-8) )
// ---------------------------------------------------------------------------
__global__ void final_kernel(const float* __restrict__ accum, float* __restrict__ out) {
    const int lane = threadIdx.x;  // 32 threads, one per batch
    float v = accum[lane] / fmaxf(accum[B_ + lane], 1e-8f);
#pragma unroll
    for (int off = 16; off > 0; off >>= 1)
        v += __shfl_xor(v, off, 32);
    if (lane == 0) out[0] = v / (float)B_;
}

// ---------------------------------------------------------------------------
extern "C" void kernel_launch(void* const* d_in, const int* in_sizes, int n_in,
                              void* d_out, int out_size, void* d_ws, size_t ws_size,
                              hipStream_t stream) {
    const float* P    = (const float*)d_in[0];
    const float* d_hw = (const float*)d_in[1];
    const int*   esrc = (const int*)d_in[2];
    const int*   edst = (const int*)d_in[3];
    const float* ew   = (const float*)d_in[4];
    float* out = (float*)d_out;

    char* ws = (char*)d_ws;
    _Float16* Dswap = (_Float16*)ws;                                  // 512 KB
    _Float16* Tmat  = (_Float16*)(ws + (size_t)Q_ * Q_ * 2);          // 16 MB
    float*    accum = (float*)(ws + (size_t)Q_ * Q_ * 2
                                  + (size_t)B_ * N_ * Q_ * 2);        // 2*B f32

    prep_kernel<<<(Q_ * Q_ + 255) / 256, 256, 0, stream>>>(d_hw, Dswap, accum);

    dim3 g2(Q_ / 64, N_ / 64, B_);
    gemm_kernel<<<g2, 128, 0, stream>>>(P, Dswap, Tmat);

    edge_kernel<<<B_ * (E_ / 8), 256, 0, stream>>>(Tmat, P, esrc, edst, ew, accum);

    final_kernel<<<1, 32, 0, stream>>>(accum, out);
}
